// Point_MA2E_PointNetv2_36764920054490
// MI455X (gfx1250) — compile-verified, matrix operand order checked by reading, not run on hardware
//
#include <hip/hip_runtime.h>
#include <math.h>

typedef __attribute__((ext_vector_type(16))) _Float16 v16h;
typedef __attribute__((ext_vector_type(8)))  float    v8f;

#define B_    8
#define N_    8192
#define NC_   64
#define GS_   32
#define GP_   36
#define BG_   (B_*NC_)       // 512
#define FM_   (BG_*GP_)      // 18432 folding rows
#define KPAD_ 1056           // 1026/1027 padded up to multiple of 32
#define LSTR_ 40             // LDS row stride in f16 (80B: 16B-aligned, bank-dephased)

// ---------- order-preserving float<->u32 (for deterministic atomic max-pool) ----------
__device__ __forceinline__ unsigned ford(float x){
  unsigned u = __float_as_uint(x);
  return (u & 0x80000000u) ? ~u : (u | 0x80000000u);
}
__device__ __forceinline__ float funord(unsigned u){
  return __uint_as_float((u & 0x80000000u) ? (u & 0x7fffffffu) : ~u);
}

// ---------- small utility kernels ----------
__global__ void zero_u32_kernel(unsigned* __restrict__ p, int n){
  int i = blockIdx.x*blockDim.x + threadIdx.x;
  if (i < n) p[i] = 0u;
}

__global__ void pack_w16_kernel(const float* __restrict__ src, _Float16* __restrict__ dst,
                                int Kin, int Kpad, int N){
  int i = blockIdx.x*blockDim.x + threadIdx.x;
  if (i >= Kpad*N) return;
  int k = i / N, n = i - k*N;
  dst[i] = (k < Kin) ? (_Float16)src[k*N + n] : (_Float16)0.f;
}

// first MLP layer from 3-D coords (act: 1=relu, 2=exact gelu), f16 output
__global__ void mlp3_kernel(const float* __restrict__ X, const float* __restrict__ W,
                            const float* __restrict__ bias, _Float16* __restrict__ out,
                            int rows, int C, int act){
  int i = blockIdx.x*blockDim.x + threadIdx.x;
  if (i >= rows*C) return;
  int r = i / C, c = i - r*C;
  float s = X[r*3]*W[c] + X[r*3+1]*W[C+c] + X[r*3+2]*W[2*C+c] + bias[c];
  if (act == 1)      s = fmaxf(s, 0.f);
  else if (act == 2) s = 0.5f*s*(1.f + erff(s*0.70710678118654752f));
  out[(size_t)r*C + c] = (_Float16)s;
}

// ---------- farthest point sampling: one block per batch ----------
__global__ void fps_kernel(const float* __restrict__ pts, float* __restrict__ centers){
  int b = blockIdx.x;
  const float* P = pts + (size_t)b*N_*3;
  __shared__ float sdist[N_];
  __shared__ float rv[256];
  __shared__ int   ri[256];
  __shared__ int   snxt;
  int t = threadIdx.x;
  float x0 = P[0], y0 = P[1], z0 = P[2];
  if (t == 0){ centers[b*NC_*3+0]=x0; centers[b*NC_*3+1]=y0; centers[b*NC_*3+2]=z0; }
  for (int i = t; i < N_; i += 256){
    float dx=P[3*i]-x0, dy=P[3*i+1]-y0, dz=P[3*i+2]-z0;
    sdist[i] = dx*dx + dy*dy + dz*dz;
  }
  __syncthreads();
  for (int s = 1; s < NC_; s++){
    float bv = -1.f; int bi = 0;
    for (int i = t; i < N_; i += 256){
      float d = sdist[i];
      if (d > bv){ bv = d; bi = i; }
    }
    rv[t] = bv; ri[t] = bi; __syncthreads();
    for (int off = 128; off > 0; off >>= 1){
      if (t < off){
        if (rv[t+off] > rv[t] || (rv[t+off] == rv[t] && ri[t+off] < ri[t])){
          rv[t] = rv[t+off]; ri[t] = ri[t+off];
        }
      }
      __syncthreads();
    }
    if (t == 0){
      int c = ri[0]; snxt = c;
      centers[(b*NC_+s)*3+0] = P[3*c];
      centers[(b*NC_+s)*3+1] = P[3*c+1];
      centers[(b*NC_+s)*3+2] = P[3*c+2];
    }
    __syncthreads();
    int nxt = snxt;
    float cx=P[3*nxt], cy=P[3*nxt+1], cz=P[3*nxt+2];
    for (int i = t; i < N_; i += 256){
      float dx=P[3*i]-cx, dy=P[3*i+1]-cy, dz=P[3*i+2]-cz;
      sdist[i] = fminf(sdist[i], dx*dx + dy*dy + dz*dz);
    }
    __syncthreads();
  }
}

// ---------- 32-NN grouping (relative coords), one block per (batch,center) ----------
__global__ void knn_kernel(const float* __restrict__ pts, const float* __restrict__ centers,
                           float* __restrict__ neigh){
  int bg = blockIdx.x;
  int b  = bg >> 6;
  const float* P = pts + (size_t)b*N_*3;
  float cx=centers[bg*3], cy=centers[bg*3+1], cz=centers[bg*3+2];
  __shared__ float sdist[N_];
  __shared__ float rv[256];
  __shared__ int   ri[256];
  int t = threadIdx.x;
  for (int i = t; i < N_; i += 256){
    float dx=P[3*i]-cx, dy=P[3*i+1]-cy, dz=P[3*i+2]-cz;
    sdist[i] = dx*dx + dy*dy + dz*dz;
  }
  __syncthreads();
  for (int s = 0; s < GS_; s++){
    float bv = 3.5e38f; int bi = 0;
    for (int i = t; i < N_; i += 256){
      float d = sdist[i];
      if (d < bv){ bv = d; bi = i; }
    }
    rv[t] = bv; ri[t] = bi; __syncthreads();
    for (int off = 128; off > 0; off >>= 1){
      if (t < off){
        if (rv[t+off] < rv[t] || (rv[t+off] == rv[t] && ri[t+off] < ri[t])){
          rv[t] = rv[t+off]; ri[t] = ri[t+off];
        }
      }
      __syncthreads();
    }
    if (t == 0){
      int c = ri[0];
      neigh[(bg*GS_+s)*3+0] = P[3*c]   - cx;
      neigh[(bg*GS_+s)*3+1] = P[3*c+1] - cy;
      neigh[(bg*GS_+s)*3+2] = P[3*c+2] - cz;
      sdist[c] = 3.4e38f;
    }
    __syncthreads();
  }
}

// ---------- WMMA GEMM: C = act(A(MxK,f16) @ B(KxN,f16) + bias) ----------
// 256 thr = 8 waves; block tile 64x128; wave tile 16x64 (A fragment reused 4x).
// Double-buffered LDS: 1 barrier per K-step; global loads of panel k+1 overlap WMMA of k.
// All five fragments are loaded before the first WMMA so the backend issues the
// ds_load_b128 batch, waits once, then fires 4 independent WMMAs back-to-back.
// MODE 0: store f16   MODE 1: store f32   MODE 2: fused max-pool via ordered atomicMax
template<int ACT, int MODE>
__global__ void __launch_bounds__(256)
wmma_gemm_kernel(const _Float16* __restrict__ A, const _Float16* __restrict__ Bm,
                 const float* __restrict__ bias,
                 _Float16* __restrict__ C16, float* __restrict__ C32,
                 unsigned* __restrict__ featU,
                 int M, int N, int K)
{
  __shared__ __align__(16) _Float16 As[2][64*LSTR_];    // A panels: 64 rows x 32 k
  __shared__ __align__(16) _Float16 Bt[2][128*LSTR_];   // B panels transposed: Bt[n][k]
  const int tid  = threadIdx.x;
  const int wave = tid >> 5;
  const int lane = tid & 31;
  const int wm = wave >> 1;          // 0..3 : 16-row slice
  const int wn = wave & 1;           // 0..1 : 64-col slice
  const int m0 = blockIdx.y * 64;
  const int n0 = blockIdx.x * 128;
  const int r  = lane & 15;
  const int hh = lane >> 4;

  v8f zero = {};
  v8f acc0 = zero, acc1 = zero, acc2 = zero, acc3 = zero;

  const int arow = tid >> 2;         // 0..63
  const int acg  = (tid & 3) << 3;   // 0,8,16,24
  const int brow = tid >> 3;         // 0..31
  const int bcg  = (tid & 7) << 4;   // 0,16,...,112

  const int nk = K >> 5;

  // per-wave LDS base offsets (in f16 elements)
  const int aBase = (wm*16 + r)*LSTR_;
  const int bBase0 = (wn*64 +  0 + r)*LSTR_;
  const int bBase1 = (wn*64 + 16 + r)*LSTR_;
  const int bBase2 = (wn*64 + 32 + r)*LSTR_;
  const int bBase3 = (wn*64 + 48 + r)*LSTR_;

  // ---- prologue: stage panel 0 ----
  {
    uint4 av = make_uint4(0,0,0,0);
    int grow = m0 + arow;
    if (grow < M) av = *(const uint4*)(A + (size_t)grow*K + acg);
    *(uint4*)&As[0][arow*LSTR_ + acg] = av;

    _Float16 btmp[16];
    const _Float16* src = Bm + (size_t)brow*N + (n0 + bcg);
    if (n0 + bcg + 16 <= N){
      *(uint4*)&btmp[0] = *(const uint4*)src;
      *(uint4*)&btmp[8] = *(const uint4*)(src + 8);
    } else {
      #pragma unroll
      for (int i = 0; i < 16; i++){
        int gc = n0 + bcg + i;
        btmp[i] = (gc < N) ? src[i] : (_Float16)0.f;
      }
    }
    #pragma unroll
    for (int i = 0; i < 16; i++) Bt[0][(bcg+i)*LSTR_ + brow] = btmp[i];
  }
  __syncthreads();

  for (int kt = 0; kt < nk; ++kt){
    const int cur = kt & 1;
    const int nxt = cur ^ 1;
    const bool have = (kt + 1) < nk;
    const int k1 = (kt + 1) << 5;

    // ---- fetch panel k+1 into registers (overlaps WMMA below) ----
    uint4 aReg = make_uint4(0,0,0,0);
    _Float16 bReg[16];
    if (have){
      int grow = m0 + arow;
      if (grow < M) aReg = *(const uint4*)(A + (size_t)grow*K + k1 + acg);
      const _Float16* src = Bm + (size_t)(k1 + brow)*N + (n0 + bcg);
      if (n0 + bcg + 16 <= N){
        *(uint4*)&bReg[0] = *(const uint4*)src;
        *(uint4*)&bReg[8] = *(const uint4*)(src + 8);
      } else {
        #pragma unroll
        for (int i = 0; i < 16; i++){
          int gc = n0 + bcg + i;
          bReg[i] = (gc < N) ? src[i] : (_Float16)0.f;
        }
      }
      if (k1 + 32 < K){ // prefetch panel k+2 (global_prefetch_b8)
        if (grow < M) __builtin_prefetch(A + (size_t)grow*K + k1 + 32 + acg, 0, 0);
        __builtin_prefetch(Bm + (size_t)(k1 + 32 + brow)*N + (n0 + bcg), 0, 0);
      }
    }

    // ---- load ALL fragments first (batched ds_loads, single wait), then 4 WMMAs ----
    v16h a, b0, b1, b2, b3;
    #pragma unroll
    for (int j = 0; j < 8; j++){
      const int kk = ((j & 4) << 2) + (hh << 3) + ((j & 3) << 1);
      a [2*j] = As[cur][aBase  + kk];  a [2*j+1] = As[cur][aBase  + kk + 1];
      b0[2*j] = Bt[cur][bBase0 + kk];  b0[2*j+1] = Bt[cur][bBase0 + kk + 1];
      b1[2*j] = Bt[cur][bBase1 + kk];  b1[2*j+1] = Bt[cur][bBase1 + kk + 1];
      b2[2*j] = Bt[cur][bBase2 + kk];  b2[2*j+1] = Bt[cur][bBase2 + kk + 1];
      b3[2*j] = Bt[cur][bBase3 + kk];  b3[2*j+1] = Bt[cur][bBase3 + kk + 1];
    }
    acc0 = __builtin_amdgcn_wmma_f32_16x16x32_f16(false, a, false, b0, (short)0, acc0, false, false);
    acc1 = __builtin_amdgcn_wmma_f32_16x16x32_f16(false, a, false, b1, (short)0, acc1, false, false);
    acc2 = __builtin_amdgcn_wmma_f32_16x16x32_f16(false, a, false, b2, (short)0, acc2, false, false);
    acc3 = __builtin_amdgcn_wmma_f32_16x16x32_f16(false, a, false, b3, (short)0, acc3, false, false);

    // ---- commit panel k+1 to the other LDS stage ----
    if (have){
      *(uint4*)&As[nxt][arow*LSTR_ + acg] = aReg;
      #pragma unroll
      for (int i = 0; i < 16; i++) Bt[nxt][(bcg+i)*LSTR_ + brow] = bReg[i];
    }
    __syncthreads();
  }

  // ---- epilogue ----
  v8f accs[4];
  accs[0] = acc0; accs[1] = acc1; accs[2] = acc2; accs[3] = acc3;
  if (MODE == 2){
    // fused per-batch column max (encoder L3): all 64 rows of this tile share one batch
    const int batch = m0 >> 13;  // 8192 rows per batch
    #pragma unroll
    for (int nt = 0; nt < 4; nt++){
      const int col = n0 + wn*64 + nt*16 + r;
      if (col < N){
        const float bb = bias[col];
        float mv = -3.4e38f;
        #pragma unroll
        for (int j = 0; j < 8; j++) mv = fmaxf(mv, accs[nt][j] + bb);
        atomicMax(&featU[batch*N + col], ford(mv));
      }
    }
  } else {
    #pragma unroll
    for (int nt = 0; nt < 4; nt++){
      const int col = n0 + wn*64 + nt*16 + r;
      if (col < N){
        const float bb = bias[col];
        #pragma unroll
        for (int j = 0; j < 8; j++){
          const int row = m0 + wm*16 + hh*8 + j;   // C/D: VGPR j -> M row (+8 upper lanes)
          if (row < M){
            float v = accs[nt][j] + bb;
            if (ACT == 1) v = fmaxf(v, 0.f);
            if (MODE == 0) C16[(size_t)row*N + col] = (_Float16)v;
            else           C32[(size_t)row*N + col] = v;
          }
        }
      }
    }
  }
}

// ---------- glue kernels ----------
__global__ void decode_feat_kernel(const unsigned* __restrict__ u, float* __restrict__ f, int n){
  int i = blockIdx.x*blockDim.x + threadIdx.x;
  if (i < n) f[i] = funord(u[i]);
}

__global__ void fe16_kernel(const float* __restrict__ feat, const float* __restrict__ pos,
                            _Float16* __restrict__ fe){
  int i = blockIdx.x*blockDim.x + threadIdx.x;     // BG_*1024
  if (i >= BG_*1024) return;
  int b = i >> 16;                                 // 64 centers * 1024 per batch
  fe[i] = (_Float16)(feat[(b << 10) + (i & 1023)] + pos[i]);
}

__global__ void cpa_kernel(const float* __restrict__ feat, _Float16* __restrict__ cpA){
  int i = blockIdx.x*blockDim.x + threadIdx.x;     // 16*1024 (rows 8..15 zero pad)
  if (i >= 16*1024) return;
  cpA[i] = (i < 8*1024) ? (_Float16)feat[i] : (_Float16)0.f;
}

__global__ void cat1_kernel(const _Float16* __restrict__ fe, _Float16* __restrict__ cat){
  int i = blockIdx.x*blockDim.x + threadIdx.x;
  if (i >= FM_*KPAD_) return;
  int rI = i / KPAD_, c = i - rI*KPAD_;
  int bg = rI / GP_,  s = rI - bg*GP_;
  _Float16 v = (_Float16)0.f;
  if (c < 1024)       v = fe[bg*1024 + c];
  else if (c == 1024) v = (_Float16)(-0.3f + 0.12f*(float)(s / 6));  // linspace(-0.3,0.3,6) 'ij'
  else if (c == 1025) v = (_Float16)(-0.3f + 0.12f*(float)(s % 6));
  cat[i] = v;
}

__global__ void cat2_kernel(const _Float16* __restrict__ fe, const float* __restrict__ fold1,
                            _Float16* __restrict__ cat){
  int i = blockIdx.x*blockDim.x + threadIdx.x;
  if (i >= FM_*KPAD_) return;
  int rI = i / KPAD_, c = i - rI*KPAD_;
  int bg = rI / GP_;
  _Float16 v = (_Float16)0.f;
  if (c < 1024)      v = fe[bg*1024 + c];
  else if (c < 1027) v = (_Float16)fold1[rI*3 + (c - 1024)];
  cat[i] = v;
}

// last folding layer: 512 -> 3 (too thin for WMMA; VALU dot)
__global__ void final3_kernel(const _Float16* __restrict__ H, const float* __restrict__ W,
                              const float* __restrict__ bias, float* __restrict__ out, int rows){
  int i = blockIdx.x*blockDim.x + threadIdx.x;
  if (i >= rows*3) return;
  int r = i / 3, o = i - r*3;
  const _Float16* h = H + (size_t)r*512;
  float s = bias[o];
  for (int k = 0; k < 512; k++) s += (float)h[k] * W[k*3 + o];
  out[i] = s;
}

// ---------- chamfer losses ----------
__global__ void chamfer_coarse_kernel(const float* __restrict__ coarseBuf,
                                      const float* __restrict__ centers,
                                      float* __restrict__ d_out){
  __shared__ float red[512];
  int t = threadIdx.x;
  int b = t >> 6, i = t & 63;
  const float* C = coarseBuf + b*192;   // (64,3) rows 0..7 of (16,192)
  const float* X = centers   + b*192;
  float ax=C[i*3], ay=C[i*3+1], az=C[i*3+2];
  float m1 = 3.4e38f;
  for (int j = 0; j < 64; j++){
    float dx=ax-X[j*3], dy=ay-X[j*3+1], dz=az-X[j*3+2];
    m1 = fminf(m1, dx*dx + dy*dy + dz*dz);
  }
  float bx=X[i*3], by=X[i*3+1], bz=X[i*3+2];
  float m2 = 3.4e38f;
  for (int j = 0; j < 64; j++){
    float dx=bx-C[j*3], dy=by-C[j*3+1], dz=bz-C[j*3+2];
    m2 = fminf(m2, dx*dx + dy*dy + dz*dz);
  }
  red[t] = m1 + m2;
  __syncthreads();
  for (int off = 256; off > 0; off >>= 1){ if (t < off) red[t] += red[t+off]; __syncthreads(); }
  if (t == 0) d_out[1] = red[0] * (1.f/512.f);
}

__global__ void chamfer_fine_partial_kernel(const float* __restrict__ fold2,
                                            const float* __restrict__ neigh,
                                            float* __restrict__ partial){
  __shared__ float F[GP_*3];
  __shared__ float G[GS_*3];
  __shared__ float red[128];
  int bg = blockIdx.x, t = threadIdx.x;
  if (t < GP_*3) F[t] = fold2[bg*GP_*3 + t];
  if (t < GS_*3) G[t] = neigh[bg*GS_*3 + t];
  __syncthreads();
  float sA = 0.f, sB = 0.f;
  if (t < GP_){
    float ax=F[t*3], ay=F[t*3+1], az=F[t*3+2];
    float m = 3.4e38f;
    for (int j = 0; j < GS_; j++){
      float dx=ax-G[j*3], dy=ay-G[j*3+1], dz=az-G[j*3+2];
      m = fminf(m, dx*dx + dy*dy + dz*dz);
    }
    sA = m;
  }
  if (t < GS_){
    float ax=G[t*3], ay=G[t*3+1], az=G[t*3+2];
    float m = 3.4e38f;
    for (int j = 0; j < GP_; j++){
      float dx=ax-F[j*3], dy=ay-F[j*3+1], dz=az-F[j*3+2];
      m = fminf(m, dx*dx + dy*dy + dz*dz);
    }
    sB = m;
  }
  red[t] = sA; __syncthreads();
  for (int off = 64; off > 0; off >>= 1){ if (t < off) red[t] += red[t+off]; __syncthreads(); }
  if (t == 0) partial[bg*2] = red[0];
  __syncthreads();
  red[t] = sB; __syncthreads();
  for (int off = 64; off > 0; off >>= 1){ if (t < off) red[t] += red[t+off]; __syncthreads(); }
  if (t == 0) partial[bg*2+1] = red[0];
}

__global__ void chamfer_fine_final_kernel(const float* __restrict__ partial, float* __restrict__ d_out){
  __shared__ float red[512];
  int t = threadIdx.x;
  red[t] = partial[t*2]; __syncthreads();
  for (int off = 256; off > 0; off >>= 1){ if (t < off) red[t] += red[t+off]; __syncthreads(); }
  float A = red[0]; __syncthreads();
  red[t] = partial[t*2+1]; __syncthreads();
  for (int off = 256; off > 0; off >>= 1){ if (t < off) red[t] += red[t+off]; __syncthreads(); }
  if (t == 0) d_out[0] = A*(1.f/18432.f) + red[0]*(1.f/16384.f);
}

// ---------------------------------------------------------------------------
extern "C" void kernel_launch(void* const* d_in, const int* in_sizes, int n_in,
                              void* d_out, int out_size, void* d_ws, size_t ws_size,
                              hipStream_t stream)
{
  (void)in_sizes; (void)n_in; (void)out_size; (void)ws_size;
  const float* corrupted = (const float*)d_in[0];
  const float* pts       = (const float*)d_in[1];
  const float* enc_w1 = (const float*)d_in[2];   const float* enc_b1 = (const float*)d_in[3];
  const float* enc_w2 = (const float*)d_in[4];   const float* enc_b2 = (const float*)d_in[5];
  const float* enc_w3 = (const float*)d_in[6];   const float* enc_b3 = (const float*)d_in[7];
  const float* pe_w1  = (const float*)d_in[8];   const float* pe_b1  = (const float*)d_in[9];
  const float* pe_w2  = (const float*)d_in[10];  const float* pe_b2  = (const float*)d_in[11];
  const float* cp_w1  = (const float*)d_in[12];  const float* cp_b1  = (const float*)d_in[13];
  const float* cp_w2  = (const float*)d_in[14];  const float* cp_b2  = (const float*)d_in[15];
  const float* cp_w3  = (const float*)d_in[16];  const float* cp_b3  = (const float*)d_in[17];
  const float* f1_w1  = (const float*)d_in[18];  const float* f1_b1  = (const float*)d_in[19];
  const float* f1_w2  = (const float*)d_in[20];  const float* f1_b2  = (const float*)d_in[21];
  const float* f1_w3  = (const float*)d_in[22];  const float* f1_b3  = (const float*)d_in[23];
  const float* f2_w1  = (const float*)d_in[24];  const float* f2_b1  = (const float*)d_in[25];
  const float* f2_w2  = (const float*)d_in[26];  const float* f2_b2  = (const float*)d_in[27];
  const float* f2_w3  = (const float*)d_in[28];  const float* f2_b3  = (const float*)d_in[29];
  float* out = (float*)d_out;

  // ---- workspace layout: encoder arena (h1+h2, 50MB) is reused for the cat buffer ----
  char* base = (char*)d_ws;
  const size_t SZ_H1 = (size_t)65536*128*2;   // 16 MB
  const size_t SZ_H2 = (size_t)65536*256*2;   // 32 MB
  _Float16* h1  = (_Float16*)base;
  _Float16* h2  = (_Float16*)(base + SZ_H1);
  _Float16* cat = (_Float16*)base;            // 38.9 MB, used only after encoder done
  size_t off = SZ_H1 + SZ_H2;
  auto alloc = [&](size_t bytes)->void*{
    off = (off + 255) & ~(size_t)255;
    void* p = (void*)(base + off);
    off += bytes;
    return p;
  };
  _Float16* hA     = (_Float16*)alloc((size_t)FM_*512*2);
  _Float16* hB     = (_Float16*)alloc((size_t)FM_*512*2);
  _Float16* w_enc2 = (_Float16*)alloc((size_t)128*256*2);
  _Float16* w_enc3 = (_Float16*)alloc((size_t)256*1024*2);
  _Float16* w_pe2  = (_Float16*)alloc((size_t)128*1024*2);
  _Float16* w_cp1  = (_Float16*)alloc((size_t)1024*1024*2);
  _Float16* w_cp2  = (_Float16*)alloc((size_t)1024*1024*2);
  _Float16* w_cp3  = (_Float16*)alloc((size_t)1024*192*2);
  _Float16* w_f11  = (_Float16*)alloc((size_t)KPAD_*512*2);
  _Float16* w_f12  = (_Float16*)alloc((size_t)512*512*2);
  _Float16* w_f21  = (_Float16*)alloc((size_t)KPAD_*512*2);
  _Float16* w_f22  = (_Float16*)alloc((size_t)512*512*2);
  float*    centers  = (float*)alloc((size_t)B_*NC_*3*4);
  float*    neigh    = (float*)alloc((size_t)BG_*GS_*3*4);
  unsigned* featU    = (unsigned*)alloc((size_t)B_*1024*4);
  float*    feat     = (float*)alloc((size_t)B_*1024*4);
  _Float16* peA      = (_Float16*)alloc((size_t)BG_*128*2);
  float*    pos      = (float*)alloc((size_t)BG_*1024*4);
  _Float16* fe16     = (_Float16*)alloc((size_t)BG_*1024*2);
  _Float16* cpA      = (_Float16*)alloc((size_t)16*1024*2);
  _Float16* cpH1     = (_Float16*)alloc((size_t)16*1024*2);
  _Float16* cpH2     = (_Float16*)alloc((size_t)16*1024*2);
  float*    coarseB  = (float*)alloc((size_t)16*192*4);
  float*    fold1    = (float*)alloc((size_t)FM_*3*4);
  float*    fold2    = (float*)alloc((size_t)FM_*3*4);
  float*    partial  = (float*)alloc((size_t)BG_*2*4);

  auto g1 = [](long long n){ return dim3((unsigned)((n + 255) / 256)); };
  auto gg = [](int M, int N){ return dim3((unsigned)((N + 127)/128), (unsigned)((M + 63)/64)); };

  // init (featU must be zero every call -> deterministic)
  zero_u32_kernel<<<g1(B_*1024), 256, 0, stream>>>(featU, B_*1024);

  // pack weights to f16 (zero-pad K to multiple of 32 where needed)
  pack_w16_kernel<<<g1(128*256),   256, 0, stream>>>(enc_w2, w_enc2, 128, 128, 256);
  pack_w16_kernel<<<g1(256*1024),  256, 0, stream>>>(enc_w3, w_enc3, 256, 256, 1024);
  pack_w16_kernel<<<g1(128*1024),  256, 0, stream>>>(pe_w2,  w_pe2,  128, 128, 1024);
  pack_w16_kernel<<<g1(1024*1024), 256, 0, stream>>>(cp_w1,  w_cp1,  1024, 1024, 1024);
  pack_w16_kernel<<<g1(1024*1024), 256, 0, stream>>>(cp_w2,  w_cp2,  1024, 1024, 1024);
  pack_w16_kernel<<<g1(1024*192),  256, 0, stream>>>(cp_w3,  w_cp3,  1024, 1024, 192);
  pack_w16_kernel<<<g1((long long)KPAD_*512), 256, 0, stream>>>(f1_w1, w_f11, 1026, KPAD_, 512);
  pack_w16_kernel<<<g1(512*512),   256, 0, stream>>>(f1_w2,  w_f12,  512, 512, 512);
  pack_w16_kernel<<<g1((long long)KPAD_*512), 256, 0, stream>>>(f2_w1, w_f21, 1027, KPAD_, 512);
  pack_w16_kernel<<<g1(512*512),   256, 0, stream>>>(f2_w2,  w_f22,  512, 512, 512);

  // geometry
  fps_kernel<<<B_,  256, 0, stream>>>(pts, centers);
  knn_kernel<<<BG_, 256, 0, stream>>>(pts, centers, neigh);

  // encoder: (65536,3)->128 relu ; WMMA 128->256 relu ; WMMA 256->1024 + fused max-pool
  mlp3_kernel<<<g1((long long)65536*128), 256, 0, stream>>>(corrupted, enc_w1, enc_b1, h1, 65536, 128, 1);
  wmma_gemm_kernel<1,0><<<gg(65536, 256), 256, 0, stream>>>(h1, w_enc2, enc_b2, h2, nullptr, nullptr, 65536, 256, 128);
  wmma_gemm_kernel<0,2><<<gg(65536, 1024), 256, 0, stream>>>(h2, w_enc3, enc_b3, nullptr, nullptr, featU, 65536, 1024, 256);
  decode_feat_kernel<<<g1(B_*1024), 256, 0, stream>>>(featU, feat, B_*1024);

  // positional embedding: gelu(centers@pe_w1+b1) ; WMMA 128->1024 (f32 out)
  mlp3_kernel<<<g1((long long)BG_*128), 256, 0, stream>>>(centers, pe_w1, pe_b1, peA, BG_, 128, 2);
  wmma_gemm_kernel<0,1><<<gg(BG_, 1024), 256, 0, stream>>>(peA, w_pe2, pe_b2, nullptr, pos, nullptr, BG_, 1024, 128);
  fe16_kernel<<<g1((long long)BG_*1024), 256, 0, stream>>>(feat, pos, fe16);

  // coarse head (M padded 8->16)
  cpa_kernel<<<g1(16*1024), 256, 0, stream>>>(feat, cpA);
  wmma_gemm_kernel<1,0><<<gg(16, 1024), 256, 0, stream>>>(cpA,  w_cp1, cp_b1, cpH1, nullptr, nullptr, 16, 1024, 1024);
  wmma_gemm_kernel<1,0><<<gg(16, 1024), 256, 0, stream>>>(cpH1, w_cp2, cp_b2, cpH2, nullptr, nullptr, 16, 1024, 1024);
  wmma_gemm_kernel<0,1><<<gg(16, 192),  256, 0, stream>>>(cpH2, w_cp3, cp_b3, nullptr, coarseB, nullptr, 16, 192, 1024);

  // folding stage 1
  cat1_kernel<<<g1((long long)FM_*KPAD_), 256, 0, stream>>>(fe16, cat);
  wmma_gemm_kernel<1,0><<<gg(FM_, 512), 256, 0, stream>>>(cat, w_f11, f1_b1, hA, nullptr, nullptr, FM_, 512, KPAD_);
  wmma_gemm_kernel<1,0><<<gg(FM_, 512), 256, 0, stream>>>(hA,  w_f12, f1_b2, hB, nullptr, nullptr, FM_, 512, 512);
  final3_kernel<<<g1((long long)FM_*3), 256, 0, stream>>>(hB, f1_w3, f1_b3, fold1, FM_);

  // folding stage 2
  cat2_kernel<<<g1((long long)FM_*KPAD_), 256, 0, stream>>>(fe16, fold1, cat);
  wmma_gemm_kernel<1,0><<<gg(FM_, 512), 256, 0, stream>>>(cat, w_f21, f2_b1, hA, nullptr, nullptr, FM_, 512, KPAD_);
  wmma_gemm_kernel<1,0><<<gg(FM_, 512), 256, 0, stream>>>(hA,  w_f22, f2_b2, hB, nullptr, nullptr, FM_, 512, 512);
  final3_kernel<<<g1((long long)FM_*3), 256, 0, stream>>>(hB, f2_w3, f2_b3, fold2, FM_);

  // losses -> d_out[0]=loss_fine, d_out[1]=loss_coarse
  chamfer_coarse_kernel<<<1, 512, 0, stream>>>(coarseB, centers, out);
  chamfer_fine_partial_kernel<<<BG_, 128, 0, stream>>>(fold2, neigh, partial);
  chamfer_fine_final_kernel<<<1, 512, 0, stream>>>(partial, out);
}
